// CEM_42777874268308
// MI455X (gfx1250) — matrix operand order repaired and checked by reading, not compiled, not run on hardware
//
#include <hip/hip_runtime.h>
#include <hip/hip_bf16.h>

// ---------------------------------------------------------------------------
// EMANet-style EM attention block on gfx1250 (wave32, WMMA bf16 16x16x32).
// GEMM operands are laid out so every WMMA fragment except mu-update's B (strided gather) is loaded as
// contiguous 128-bit vector loads per lane (A: row-major source, B:
// column-major source). bf16 storage keeps the EM working set (~95MB)
// resident in the 192MB L2.
// ---------------------------------------------------------------------------

typedef __attribute__((ext_vector_type(16))) __bf16 v16bf;
typedef __attribute__((ext_vector_type(8)))  float  v8f;

#define DEV static __device__ __forceinline__

// Problem dims
#define Bb    8
#define Nimg  5
#define NIMG  40          // Bb*Nimg
#define Cc    512
#define Kk    64
#define HWp   1024        // 32*32
#define Nn    5120        // Nimg*HWp
#define STAGES 5

DEV v8f wmma_bf16(v16bf a, v16bf b, v8f c) {
  return __builtin_amdgcn_wmma_f32_16x16x32_bf16(false, a, false, b, (short)0, c,
                                                 false, false);
}

DEV v16bf frag2(uint4 q0, uint4 q1) {
  union { uint4 q[2]; v16bf f; } u;
  u.q[0] = q0; u.q[1] = q1;
  return u.f;
}

// A fragment (16x32 MxK), row-major bf16 source, tile origin t, ldim ld.
// Order used here (same as B): lane m=lane&15; elem e -> K = 16*hi + e  (16 contiguous K's)
DEV v16bf loadA_bf16v(const __bf16* __restrict__ t, int ld, int m, int hi) {
  const __bf16* r = t + (size_t)m * ld + 16 * hi;
  uint4 q0 = *(const uint4*)r;          // K run [16hi .. 16hi+7]
  uint4 q1 = *(const uint4*)(r + 8);    // K run [16hi+8 .. 16hi+15]
  return frag2(q0, q1);
}
// B fragment (32x16 KxN) from COLUMN-major source: src[n][k], k contiguous.
// Order used here (same as A): lane n=lane&15; elem e -> K = 16*hi + e  (16 contiguous K's)
DEV v16bf loadBc_bf16v(const __bf16* __restrict__ rowp /* = src + n*ld + kc + 16*hi */) {
  uint4 q0 = *(const uint4*)rowp;
  uint4 q1 = *(const uint4*)(rowp + 8);
  return frag2(q0, q1);
}
// Same but f32 source (16 contiguous floats, convert to bf16)
DEV v16bf loadBc_f32v(const float* __restrict__ rowp) {
  float4 f0 = ((const float4*)rowp)[0];
  float4 f1 = ((const float4*)rowp)[1];
  float4 f2 = ((const float4*)rowp)[2];
  float4 f3 = ((const float4*)rowp)[3];
  v16bf b;
  b[0]=(__bf16)f0.x;  b[1]=(__bf16)f0.y;  b[2]=(__bf16)f0.z;  b[3]=(__bf16)f0.w;
  b[4]=(__bf16)f1.x;  b[5]=(__bf16)f1.y;  b[6]=(__bf16)f1.z;  b[7]=(__bf16)f1.w;
  b[8]=(__bf16)f2.x;  b[9]=(__bf16)f2.y;  b[10]=(__bf16)f2.z; b[11]=(__bf16)f2.w;
  b[12]=(__bf16)f3.x; b[13]=(__bf16)f3.y; b[14]=(__bf16)f3.z; b[15]=(__bf16)f3.w;
  return b;
}

// ---------------------------------------------------------------------------
// generic f32 -> bf16 convert (weights)
__global__ void k_cvt_bf16(const float* __restrict__ src, __bf16* __restrict__ dst, int n) {
  int i = blockIdx.x * blockDim.x + threadIdx.x;
  if (i < n) dst[i] = (__bf16)src[i];
}

// ---------------------------------------------------------------------------
// transpose-convert x[img][c][p] f32 -> xt[img][p][c] bf16 (LDS 32x32 tiles)
__global__ void k_x2bf(const float* __restrict__ x, __bf16* __restrict__ xt) {
  __shared__ __bf16 tile[32][34];
  int blk = blockIdx.x;
  int pt = blk & 31;             // 32 pixel tiles of 32
  int ct = (blk >> 5) & 15;      // 16 channel tiles of 32
  int img = blk >> 9;            // 40 images
  int tid = threadIdx.x;
  int c0 = ct * 32, p0 = pt * 32;
  const float* xi = x + (size_t)img * Cc * HWp;
#pragma unroll
  for (int i = 0; i < 4; i++) {
    int cl = (tid >> 5) + i * 8;     // 0..31
    int pl = tid & 31;
    tile[cl][pl] = (__bf16)xi[(size_t)(c0 + cl) * HWp + p0 + pl];
  }
  __syncthreads();
  __bf16* xo = xt + (size_t)img * HWp * Cc;
  int pl = tid >> 3;               // 0..31
  int cg = (tid & 7) * 4;          // 4 contiguous channels per thread
  __bf16 o[4];
#pragma unroll
  for (int j = 0; j < 4; j++) o[j] = tile[cg + j][pl];
  *(uint2*)(xo + (size_t)(p0 + pl) * Cc + c0 + cg) = *(uint2*)o;
}

// ---------------------------------------------------------------------------
// mu_init: broadcast input mu (1,512,64) -> mubf[b][c][k] and mutbf[b][k][c]
__global__ void k_mu_init(const float* __restrict__ mu_in, __bf16* __restrict__ mubf,
                          __bf16* __restrict__ mutbf) {
  int i = blockIdx.x * blockDim.x + threadIdx.x;   // Bb*Cc*Kk threads
  int ck = i % (Cc * Kk);
  int b = i / (Cc * Kk);
  int c = ck / Kk, k = ck % Kk;
  __bf16 v = (__bf16)mu_in[ck];
  mubf[i] = v;
  mutbf[((size_t)b * Kk + k) * Cc + c] = v;
}

// ---------------------------------------------------------------------------
// conv1: y[d][p] = sum_c W1[d][c] x[c][p] + b1[d]
// writes Y  [b][c][n]  (for mu-update A reads, row-major in n)
//   and  Yt [b][n][c]  (for logits A reads, row-major in c; packed b128 store)
__global__ void k_conv1(const __bf16* __restrict__ xt, const __bf16* __restrict__ w1bf,
                        const float* __restrict__ b1, __bf16* __restrict__ ybf,
                        __bf16* __restrict__ ytbf) {
  int wave = (blockIdx.x * blockDim.x + threadIdx.x) >> 5;
  int lane = threadIdx.x & 31;
  int ln = lane & 15, hi = lane >> 4;
  int pt = wave & 63;            // 64 pixel tiles of 16
  int mt = (wave >> 6) & 31;     // 32 out-channel tiles
  int img = wave >> 11;          // 40 images
  const __bf16* xi = xt + (size_t)img * HWp * Cc;    // [p][c]
  const __bf16* brow = xi + (size_t)(pt * 16 + ln) * Cc + 16 * hi;
  v8f acc = {};
  for (int kc = 0; kc < Cc; kc += 32) {
    __builtin_prefetch(brow + kc + 64, 0, 0);
    v16bf a = loadA_bf16v(w1bf + (size_t)(mt * 16) * Cc + kc, Cc, ln, hi);
    v16bf b = loadBc_bf16v(brow + kc);
    acc = wmma_bf16(a, b, acc);
  }
  int bidx = img / Nimg, nimg = img % Nimg;
  int p = pt * 16 + ln;
  int n = nimg * HWp + p;
  __bf16 o[8];
#pragma unroll
  for (int r = 0; r < 8; r++) {
    int d = mt * 16 + hi * 8 + r;
    float v = acc[r] + b1[d];
    o[r] = (__bf16)v;
    ybf[((size_t)(bidx * Cc + d)) * Nn + n] = o[r];          // scattered b16
  }
  // Yt: 8 contiguous channels -> one b128 store
  *(uint4*)(ytbf + ((size_t)bidx * Nn + n) * Cc + mt * 16 + hi * 8) = *(uint4*)o;
}

// ---------------------------------------------------------------------------
// Z = softmax_k( Y^T mu ); accumulate colsum[b][k] = sum_n Z[b][n][k]
__global__ void k_logits_softmax(const __bf16* __restrict__ ytbf,
                                 const __bf16* __restrict__ mutbf,
                                 float* __restrict__ z, float* __restrict__ colsum) {
  int wave = (blockIdx.x * blockDim.x + threadIdx.x) >> 5;
  int lane = threadIdx.x & 31;
  int ln = lane & 15, hi = lane >> 4;
  int nt = wave % (Nn / 16);     // 320 row tiles
  int b  = wave / (Nn / 16);
  const __bf16* yt = ytbf + (size_t)b * Nn * Cc;     // [n][c]
  const __bf16* mt_ = mutbf + (size_t)b * Kk * Cc;   // [k][c]
  const __bf16* arow = yt + (size_t)(nt * 16) * Cc;
  v8f acc[4] = {{}, {}, {}, {}};
  for (int kc = 0; kc < Cc; kc += 32) {
    v16bf a = loadA_bf16v(arow + kc, Cc, ln, hi);    // A[m=n][k=c], contiguous
#pragma unroll
    for (int t = 0; t < 4; t++) {                    // B[k=c][n=kk] = mut[kk][c]
      v16bf bf = loadBc_bf16v(mt_ + (size_t)(t * 16 + ln) * Cc + kc + 16 * hi);
      acc[t] = wmma_bf16(a, bf, acc[t]);
    }
  }
  float* zb = z + (size_t)b * Nn * Kk;
#pragma unroll
  for (int r = 0; r < 8; r++) {
    float mx = -1e30f;
#pragma unroll
    for (int t = 0; t < 4; t++) mx = fmaxf(mx, acc[t][r]);
    for (int s = 1; s < 16; s <<= 1) mx = fmaxf(mx, __shfl_xor(mx, s, 32));
    float ev[4], sum = 0.f;
#pragma unroll
    for (int t = 0; t < 4; t++) { ev[t] = __expf(acc[t][r] - mx); sum += ev[t]; }
    for (int s = 1; s < 16; s <<= 1) sum += __shfl_xor(sum, s, 32);
    float inv = 1.0f / sum;
    int nrow = nt * 16 + r + hi * 8;
#pragma unroll
    for (int t = 0; t < 4; t++) {
      float zv = ev[t] * inv;
      zb[(size_t)nrow * Kk + t * 16 + ln] = zv;
      acc[t][r] = zv;
    }
  }
#pragma unroll
  for (int t = 0; t < 4; t++) {
    float part = 0.f;
#pragma unroll
    for (int r = 0; r < 8; r++) part += acc[t][r];
    part += __shfl_xor(part, 16, 32);
    if (hi == 0) atomicAdd(&colsum[b * Kk + t * 16 + ln], part);
  }
}

// ---------------------------------------------------------------------------
// mu_raw[b][c][k] = sum_n Y[b][c][n] * Z[b][n][k] * rc[k]
__global__ void k_mu_update(const __bf16* __restrict__ ybf, const float* __restrict__ z,
                            const float* __restrict__ colsum, float* __restrict__ muraw) {
  int wave = (blockIdx.x * blockDim.x + threadIdx.x) >> 5;
  int lane = threadIdx.x & 31;
  int ln = lane & 15, hi = lane >> 4;
  int kt = wave & 3;             // 4 k-tiles
  int mt = (wave >> 2) & 31;     // 32 c-tiles
  int b  = wave >> 7;            // 8 batches
  const __bf16* yb = ybf + (size_t)b * Cc * Nn;
  const float* zb = z + (size_t)b * Nn * Kk;
  int col = kt * 16 + ln;
  float rc = 1.0f / (1e-6f + colsum[b * Kk + col]);   // fixed per lane
  const __bf16* arow = yb + (size_t)(mt * 16) * Nn;
  v8f acc = {};
  for (int kn = 0; kn < Nn; kn += 32) {
    v16bf a = loadA_bf16v(arow + kn, Nn, ln, hi);
    v16bf bf;                    // B[k=n][col] = Z[n][col]*rc : k-strided gather
#pragma unroll
    for (int e = 0; e < 16; e++) {
      int kr = kn + (hi << 4) + e;
      bf[e] = (__bf16)(zb[(size_t)kr * Kk + col] * rc);
    }
    acc = wmma_bf16(a, bf, acc);
  }
#pragma unroll
  for (int r = 0; r < 8; r++) {
    int c = mt * 16 + r + hi * 8;
    muraw[((size_t)(b * Cc + c)) * Kk + col] = acc[r];
  }
}

// ---------------------------------------------------------------------------
// L2-normalize mu over c; emit f32 [c][k], bf16 [c][k], bf16-transposed [k][c].
__global__ void k_mu_norm(const float* __restrict__ muraw, float* __restrict__ mu,
                          __bf16* __restrict__ mubf, __bf16* __restrict__ mutbf) {
  int wave = (blockIdx.x * blockDim.x + threadIdx.x) >> 5;
  int lane = threadIdx.x & 31;
  int k = wave & (Kk - 1);
  int b = wave >> 6;
  const float* src = muraw + (size_t)b * Cc * Kk + k;
  float ss = 0.f;
  for (int c = lane; c < Cc; c += 32) { float v = src[(size_t)c * Kk]; ss += v * v; }
  for (int s = 1; s < 32; s <<= 1) ss += __shfl_xor(ss, s, 32);
  float inv = 1.0f / (1e-6f + sqrtf(ss));
  float* dst = mu + (size_t)b * Cc * Kk + k;
  __bf16* dbf = mubf + (size_t)b * Cc * Kk + k;
  __bf16* dt = mutbf + ((size_t)b * Kk + k) * Cc;
  for (int c = lane; c < Cc; c += 32) {
    float v = src[(size_t)c * Kk] * inv;
    dst[(size_t)c * Kk] = v;
    dbf[(size_t)c * Kk] = (__bf16)v;
    dt[c] = (__bf16)v;           // contiguous across lanes
  }
}

// ---------------------------------------------------------------------------
// rect[img][p][c] = relu( sum_k mu[b][c][k] * Z[b][n][k] ), transposed bf16
__global__ void k_rec(const __bf16* __restrict__ mubf, const float* __restrict__ z,
                      __bf16* __restrict__ rect) {
  int wave = (blockIdx.x * blockDim.x + threadIdx.x) >> 5;
  int lane = threadIdx.x & 31;
  int ln = lane & 15, hi = lane >> 4;
  int nt = wave % (Nn / 16);
  int rem = wave / (Nn / 16);
  int ct = rem & 31;
  int b  = rem >> 5;
  const __bf16* mb = mubf + (size_t)b * Cc * Kk;     // [c][k]
  const float* zb = z + (size_t)b * Nn * Kk;         // [n][k]
  int n = nt * 16 + ln;
  v8f acc = {};
#pragma unroll
  for (int kc = 0; kc < Kk; kc += 32) {
    v16bf a = loadA_bf16v(mb + (size_t)(ct * 16) * Kk + kc, Kk, ln, hi);
    v16bf bf = loadBc_f32v(zb + (size_t)n * Kk + kc + 16 * hi);  // contiguous
    acc = wmma_bf16(a, bf, acc);
  }
  int nimg = n >> 10, p = n & (HWp - 1);
  int img = b * Nimg + nimg;
  __bf16 o[8];
#pragma unroll
  for (int r = 0; r < 8; r++) o[r] = (__bf16)fmaxf(acc[r], 0.f);
  // 8 contiguous channels -> one b128 store, transposed layout [p][c]
  *(uint4*)(rect + ((size_t)img * HWp + p) * Cc + ct * 16 + hi * 8) = *(uint4*)o;
}

// ---------------------------------------------------------------------------
// conv2 tile (shared by stats pass and fused output pass); rect is [p][c]
DEV v8f conv2_tile(const __bf16* __restrict__ w2bf, const __bf16* __restrict__ rimg,
                   int dt, int pt, int ln, int hi) {
  const __bf16* brow = rimg + (size_t)(pt * 16 + ln) * Cc + 16 * hi;
  v8f acc = {};
  for (int kc = 0; kc < Cc; kc += 32) {
    __builtin_prefetch(brow + kc + 64, 0, 0);
    v16bf a = loadA_bf16v(w2bf + (size_t)(dt * 16) * Cc + kc, Cc, ln, hi);
    v16bf b = loadBc_bf16v(brow + kc);
    acc = wmma_bf16(a, b, acc);
  }
  return acc;
}

// Pass 1: per-channel sum / sumsq for training-mode BN (no y2 materialization)
__global__ void k_conv2_stats(const __bf16* __restrict__ w2bf, const __bf16* __restrict__ rect,
                              float* __restrict__ bnsum, float* __restrict__ bnsq) {
  int wave = (blockIdx.x * blockDim.x + threadIdx.x) >> 5;
  int lane = threadIdx.x & 31;
  int ln = lane & 15, hi = lane >> 4;
  int pt = wave & 63;
  int dt = (wave >> 6) & 31;
  int img = wave >> 11;
  v8f acc = conv2_tile(w2bf, rect + (size_t)img * HWp * Cc, dt, pt, ln, hi);
#pragma unroll
  for (int r = 0; r < 8; r++) {
    float s = acc[r], q = s * s;
    for (int m = 1; m < 16; m <<= 1) {
      s += __shfl_xor(s, m, 32);
      q += __shfl_xor(q, m, 32);
    }
    if (ln == 0) {
      int d = dt * 16 + r + hi * 8;
      atomicAdd(&bnsum[d], s);
      atomicAdd(&bnsq[d], q);
    }
  }
}

__global__ void k_bn_final(const float* __restrict__ bnsum, const float* __restrict__ bnsq,
                           const float* __restrict__ gamma, const float* __restrict__ beta,
                           float* __restrict__ scale, float* __restrict__ shift) {
  int d = blockIdx.x * blockDim.x + threadIdx.x;
  if (d >= Cc) return;
  const float invN = 1.0f / (float)(NIMG * HWp);
  float mean = bnsum[d] * invN;
  float var = bnsq[d] * invN - mean * mean;
  float sc = gamma[d] * rsqrtf(var + 1e-5f);
  scale[d] = sc;
  shift[d] = beta[d] - mean * sc;
}

// Pass 2: recompute conv2 tile, apply BN + residual + relu, write output
__global__ void k_conv2_fuse(const __bf16* __restrict__ w2bf, const __bf16* __restrict__ rect,
                             const float* __restrict__ scale, const float* __restrict__ shift,
                             const float* __restrict__ x, float* __restrict__ out) {
  int wave = (blockIdx.x * blockDim.x + threadIdx.x) >> 5;
  int lane = threadIdx.x & 31;
  int ln = lane & 15, hi = lane >> 4;
  int pt = wave & 63;
  int dt = (wave >> 6) & 31;
  int img = wave >> 11;
  v8f acc = conv2_tile(w2bf, rect + (size_t)img * HWp * Cc, dt, pt, ln, hi);
  int p = pt * 16 + ln;
#pragma unroll
  for (int r = 0; r < 8; r++) {
    int d = dt * 16 + r + hi * 8;
    size_t idx = ((size_t)(img * Cc + d)) * HWp + p;
    float v = acc[r] * scale[d] + shift[d] + x[idx];
    out[idx] = fmaxf(v, 0.f);
  }
}

// ---------------------------------------------------------------------------
extern "C" void kernel_launch(void* const* d_in, const int* in_sizes, int n_in,
                              void* d_out, int out_size, void* d_ws, size_t ws_size,
                              hipStream_t stream) {
  const float* x   = (const float*)d_in[0];  // (8,5,512,32,32)
  const float* mu0 = (const float*)d_in[1];  // (1,512,64)
  const float* w1  = (const float*)d_in[2];  // (512,512)
  const float* b1  = (const float*)d_in[3];  // (512,)
  const float* w2  = (const float*)d_in[4];  // (512,512)
  const float* gam = (const float*)d_in[5];  // (512,)
  const float* bet = (const float*)d_in[6];  // (512,)
  float* out = (float*)d_out;                // 8*5*512*1024 + 8*512*64 floats

  char* ws = (char*)d_ws;
  size_t off = 0;
  auto alloc = [&](size_t bytes) -> void* {
    void* p = ws + off;
    off = (off + bytes + 255) & ~(size_t)255;
    return p;
  };
  __bf16* xt    = (__bf16*)alloc((size_t)NIMG * HWp * Cc * 2);  // ~42 MB
  __bf16* ybf   = (__bf16*)alloc((size_t)Bb * Cc * Nn * 2);     // ~42 MB
  __bf16* ytbf  = (__bf16*)alloc((size_t)Bb * Nn * Cc * 2);     // ~42 MB
  float*  zbuf  = (float*)alloc((size_t)Bb * Nn * Kk * 4);      // ~10.5 MB
  __bf16* rect  = (__bf16*)alloc((size_t)NIMG * HWp * Cc * 2);  // ~42 MB
  __bf16* w1bf  = (__bf16*)alloc((size_t)Cc * Cc * 2);
  __bf16* w2bf  = (__bf16*)alloc((size_t)Cc * Cc * 2);
  __bf16* mubf  = (__bf16*)alloc((size_t)Bb * Cc * Kk * 2);
  __bf16* mutbf = (__bf16*)alloc((size_t)Bb * Kk * Cc * 2);
  float*  muf   = (float*)alloc((size_t)Bb * Cc * Kk * 4);
  float*  muraw = (float*)alloc((size_t)Bb * Cc * Kk * 4);
  float*  colsum = (float*)alloc((size_t)Bb * Kk * 4);
  float*  bnsum  = (float*)alloc(Cc * 4);
  float*  bnsq   = (float*)alloc(Cc * 4);
  float*  bnscale = (float*)alloc(Cc * 4);
  float*  bnshift = (float*)alloc(Cc * 4);

  const int BLK = 128;  // 4 waves per block

  // layout prep: weights -> bf16, x -> transposed bf16, mu broadcast
  k_cvt_bf16<<<(Cc * Cc + 255) / 256, 256, 0, stream>>>(w1, w1bf, Cc * Cc);
  k_cvt_bf16<<<(Cc * Cc + 255) / 256, 256, 0, stream>>>(w2, w2bf, Cc * Cc);
  k_x2bf<<<NIMG * 16 * 32, 256, 0, stream>>>(x, xt);
  k_mu_init<<<(Bb * Cc * Kk) / 256, 256, 0, stream>>>(mu0, mubf, mutbf);

  // conv1 -> Y [b][c][n] and Yt [b][n][c]
  k_conv1<<<(NIMG * 32 * 64) / 4, BLK, 0, stream>>>(xt, w1bf, b1, ybf, ytbf);

  // EM iterations
  for (int it = 0; it < STAGES; it++) {
    hipMemsetAsync(colsum, 0, (size_t)Bb * Kk * 4, stream);
    k_logits_softmax<<<(Bb * (Nn / 16)) / 4, BLK, 0, stream>>>(ytbf, mutbf, zbuf, colsum);
    k_mu_update<<<(Bb * 32 * 4) / 4, BLK, 0, stream>>>(ybf, zbuf, colsum, muraw);
    k_mu_norm<<<(Bb * Kk) / 4, BLK, 0, stream>>>(muraw, muf, mubf, mutbf);
  }

  // reconstruction (final mu + last-iteration z) -> rect [img][p][c]
  k_rec<<<(Bb * 32 * (Nn / 16)) / 4, BLK, 0, stream>>>(mubf, zbuf, rect);

  // conv2 + BN(train) + residual + relu: stats pass then fused apply pass
  hipMemsetAsync(bnsum, 0, Cc * 4, stream);
  hipMemsetAsync(bnsq, 0, Cc * 4, stream);
  k_conv2_stats<<<(NIMG * 32 * 64) / 4, BLK, 0, stream>>>(w2bf, rect, bnsum, bnsq);
  k_bn_final<<<(Cc + 255) / 256, 256, 0, stream>>>(bnsum, bnsq, gam, bet, bnscale, bnshift);
  k_conv2_fuse<<<(NIMG * 32 * 64) / 4, BLK, 0, stream>>>(w2bf, rect, bnscale, bnshift, x, out);

  // second output: final mu_b (8,512,64) f32 appended after the main tensor
  hipMemcpyAsync(out + (size_t)Bb * Nimg * Cc * HWp, muf,
                 (size_t)Bb * Cc * Kk * 4, hipMemcpyDeviceToDevice, stream);
}
